// QuantumManiModel_9543417331910
// MI455X (gfx1250) — compile-verified, matrix operand order checked by reading, not run on hardware
//
#include <hip/hip_runtime.h>
#include <math.h>

// ---------------------------------------------------------------------------
// Types for CDNA5 WMMA / TDM (gfx1250, wave32)
// ---------------------------------------------------------------------------
typedef __attribute__((ext_vector_type(16))) __bf16 v16bf;
typedef __attribute__((ext_vector_type(8)))  float  v8f;
typedef __attribute__((ext_vector_type(4)))  unsigned int u32x4;
typedef __attribute__((ext_vector_type(4)))  int    i32x4;
typedef __attribute__((ext_vector_type(8)))  int    i32x8;

union U16bf {
    v16bf v;
    u32x4 q[2];
};

#define BATCH 8
#define NPTS  2048
#define EMB   16
#define HEADS 2
#define HDIM  8

// ---------------------------------------------------------------------------
// Workspace layout (bytes)
// ---------------------------------------------------------------------------
static constexpr size_t WS_QMAT  = 0;                       // 16 mats * 512 f32 = 32KB
static constexpr size_t WS_X0    = WS_QMAT + 32768;         // 1 MB
static constexpr size_t WS_X1    = WS_X0   + (size_t)BATCH*NPTS*EMB*4;
static constexpr size_t WS_QB    = WS_X1   + (size_t)BATCH*NPTS*EMB*4;  // bf16 1MB
static constexpr size_t WS_KB    = WS_QB   + (size_t)BATCH*HEADS*NPTS*16*2;
static constexpr size_t WS_VB    = WS_KB   + (size_t)BATCH*HEADS*NPTS*16*2;
static constexpr size_t WS_HEAT  = WS_VB   + (size_t)BATCH*HEADS*16*NPTS*2;
static constexpr size_t WS_FEATS = WS_HEAT + (size_t)BATCH*NPTS*4;
static constexpr size_t WS_POSP  = WS_FEATS+ (size_t)BATCH*NPTS*9*4;

// ---------------------------------------------------------------------------
// TDM: async DMA of a contiguous byte range (as a 1-row 2D bf16 tensor) into
// LDS. Descriptor per CDNA5 ISA ch.8 (D# groups). Tracked by TENSORcnt.
// This toolchain exposes the 6-arg builtin:
//   (u32x4 g0, i32x8 g1, i32x4 g2, i32x4 g3, i32x8 g4, i32 cpol)
// ---------------------------------------------------------------------------
#if __has_builtin(__builtin_amdgcn_tensor_load_to_lds)
#define HAVE_TDM 1
__device__ __forceinline__ void tdm_copy_to_lds(unsigned lds_off, const void* gptr,
                                                unsigned nbytes) {
    const unsigned elems = nbytes >> 1;                  // data_size = 2 bytes
    const unsigned long long ga = (unsigned long long)gptr;
    const u32x4 g0 = {
        1u,                                              // count=1, user mode
        lds_off,                                         // lds_addr (bytes)
        (unsigned)(ga & 0xffffffffu),                    // global_addr[31:0]
        (unsigned)((ga >> 32) & 0x01ffffffu) | (2u << 30)  // addr[56:32] | type=2
    };
    const i32x8 g1 = {
        (int)0x00010000u,                                // data_size=1 (2B), mask=0
        (int)((elems & 0xffffu) << 16),                  // tensor_dim0[15:0]
        (int)(((elems >> 16) & 0xffffu) | (1u << 16)),   // tensor_dim0[31:16] | tensor_dim1=1
        (int)((elems & 0xffffu) << 16),                  // tile_dim0
        1,                                               // tile_dim1=1
        (int)elems,                                      // tensor_dim0_stride[31:0]
        0, 0
    };
    const i32x4 gz4 = {0, 0, 0, 0};
    const i32x8 gz8 = {0, 0, 0, 0, 0, 0, 0, 0};
    __builtin_amdgcn_tensor_load_to_lds(g0, g1, gz4, gz4, gz8, 0);
}
#else
#define HAVE_TDM 0
#endif

// ---------------------------------------------------------------------------
// Kernel 0: 16 complex 16x16 QR factorizations (modified Gram-Schmidt).
// One wave per matrix; lane i holds row i of every column in registers.
// Writes Q row-major: [mat][re(256) | im(256)]
// ---------------------------------------------------------------------------
struct QRArgs { const float* wr[16]; const float* wi[16]; };

__global__ __launch_bounds__(32) void qr16_kernel(QRArgs args, float* qout) {
    const int m = blockIdx.x;
    const int lane = threadIdx.x;
    if (lane >= 16) return;
    const float* wr = args.wr[m];
    const float* wi = args.wi[m];
    float cr[16], ci[16];          // this lane's row, all 16 columns
    #pragma unroll
    for (int j = 0; j < 16; ++j) { cr[j] = wr[lane*16 + j]; ci[j] = wi[lane*16 + j]; }
    #pragma unroll
    for (int j = 0; j < 16; ++j) {
        float s = cr[j]*cr[j] + ci[j]*ci[j];
        s += __shfl_xor(s, 1, 32); s += __shfl_xor(s, 2, 32);
        s += __shfl_xor(s, 4, 32); s += __shfl_xor(s, 8, 32);
        float inv = rsqrtf(fmaxf(s, 1e-30f));
        float qjr = cr[j] * inv, qji = ci[j] * inv;
        cr[j] = qjr; ci[j] = qji;
        #pragma unroll
        for (int k = j + 1; k < 16; ++k) {
            float xr = cr[k], xi = ci[k];
            float pr = qjr*xr + qji*xi;      // Re(conj(qj) . ck)
            float pi = qjr*xi - qji*xr;      // Im
            pr += __shfl_xor(pr, 1, 32); pr += __shfl_xor(pr, 2, 32);
            pr += __shfl_xor(pr, 4, 32); pr += __shfl_xor(pr, 8, 32);
            pi += __shfl_xor(pi, 1, 32); pi += __shfl_xor(pi, 2, 32);
            pi += __shfl_xor(pi, 4, 32); pi += __shfl_xor(pi, 8, 32);
            cr[k] = xr - (pr*qjr - pi*qji);
            ci[k] = xi - (pr*qji + pi*qjr);
        }
    }
    float* qre = qout + (size_t)m * 512;
    float* qim = qre + 256;
    #pragma unroll
    for (int c = 0; c < 16; ++c) { qre[lane*16 + c] = cr[c]; qim[lane*16 + c] = ci[c]; }
}

// ---------------------------------------------------------------------------
// Kernel 1: x = rgb @ w_in.T + b_in   -> [B,N,16] f32
// ---------------------------------------------------------------------------
__global__ __launch_bounds__(256) void lin_in_kernel(const float* __restrict__ rgb,
                                                     const float* __restrict__ w_in,
                                                     const float* __restrict__ b_in,
                                                     float* __restrict__ X) {
    const int row = blockIdx.x * 256 + threadIdx.x;       // 0..B*N-1
    const float r0 = rgb[(size_t)row*3 + 0];
    const float r1 = rgb[(size_t)row*3 + 1];
    const float r2 = rgb[(size_t)row*3 + 2];
    #pragma unroll
    for (int e = 0; e < 16; ++e) {
        X[(size_t)row*16 + e] = b_in[e] + r0*w_in[e*3] + r1*w_in[e*3+1] + r2*w_in[e*3+2];
    }
}

// ---------------------------------------------------------------------------
// Kernel 2: complex projections q,k,v = x @ conj(Q).T  (x real).
//   q,k stored row-major bf16: [b,h][n][qr0..7 | qi0..7]  (WMMA A/B friendly)
//   v   stored comp-major bf16: [b,h][comp0..15][n]       (WMMA B for P@V)
// ---------------------------------------------------------------------------
__global__ __launch_bounds__(256) void proj_qkv_kernel(const float* __restrict__ X,
                                                       const float* __restrict__ qmQ,
                                                       const float* __restrict__ qmK,
                                                       const float* __restrict__ qmV,
                                                       __bf16* __restrict__ qb,
                                                       __bf16* __restrict__ kb,
                                                       __bf16* __restrict__ vb) {
    __shared__ float sQ[3][512];
    const int tid = threadIdx.x;
    for (int i = tid; i < 512; i += 256) {
        sQ[0][i] = qmQ[i]; sQ[1][i] = qmK[i]; sQ[2][i] = qmV[i];
    }
    __syncthreads();
    const int row = blockIdx.x * 256 + tid;
    const int b = row >> 11, n = row & (NPTS - 1);
    float x[16];
    #pragma unroll
    for (int c = 0; c < 16; ++c) x[c] = X[(size_t)row*16 + c];
    for (int t = 0; t < 3; ++t) {
        for (int e = 0; e < 16; ++e) {
            float re = 0.f, im = 0.f;
            #pragma unroll
            for (int c = 0; c < 16; ++c) {
                re += x[c] * sQ[t][e*16 + c];
                im += x[c] * sQ[t][256 + e*16 + c];
            }
            im = -im;                                    // conj
            const int h = e >> 3, d = e & 7;
            const size_t bh = (size_t)(b*HEADS + h);
            if (t == 0) {
                qb[(bh*NPTS + n)*16 + d]     = (__bf16)re;
                qb[(bh*NPTS + n)*16 + 8 + d] = (__bf16)im;
            } else if (t == 1) {
                kb[(bh*NPTS + n)*16 + d]     = (__bf16)re;
                kb[(bh*NPTS + n)*16 + 8 + d] = (__bf16)im;
            } else {
                vb[(bh*16 + d)*NPTS + n]     = (__bf16)re;
                vb[(bh*16 + 8 + d)*NPTS + n] = (__bf16)im;
            }
        }
    }
}

// ---------------------------------------------------------------------------
// Kernel 3: fused flash attention + phase gate + Wo unitary + re+0.1*im.
// Grid: 256 blocks (b * 32 row-tiles of 64), 8 waves; wave -> (head, 16-row tile).
// Both heads' K panels (2 x 64KB bf16) are staged into LDS once per block by
// the Tensor Data Mover (TENSORcnt + s_wait_tensorcnt), then the inner loop
// serves score-WMMA B operands from LDS. Per 32-key chunk: 4 score WMMAs
// (Re/Im x 2 key subtiles), online softmax in the f32 C layout, P via LDS
// transpose, 1 P@V WMMA (K=32 keys, full utilization).
// ---------------------------------------------------------------------------
__global__ __launch_bounds__(256) void attn_kernel(const __bf16* __restrict__ qb,
                                                   const __bf16* __restrict__ kb,
                                                   const __bf16* __restrict__ vb,
                                                   const float* __restrict__ qo,
                                                   const float* __restrict__ theta,
                                                   const float* __restrict__ amp,
                                                   float* __restrict__ Xout) {
    __shared__ __align__(16) __bf16 sK[HEADS][NPTS*16];  // 128KB: K panels, both heads
    __shared__ __align__(16) __bf16 sP[8][16*32];        // per-wave P tile (C->A transpose)
    __shared__ float sComb[64][32];                      // merged heads: [row][re16|im16]
    __shared__ float sQr[256], sQi[256];
    __shared__ float sCos[16], sSin[16];

    const int tid  = threadIdx.x;
    const int wave = tid >> 5;
    const int lane = tid & 31;
    const int b    = blockIdx.x >> 5;
    const int tb   = blockIdx.x & 31;
    const int h    = wave & 1;
    const int qt   = wave >> 1;                     // 0..3
    const int rowBase = tb*64 + qt*16;

    // ---- async stage of K panels into LDS (TDM), plus small constant staging
#if HAVE_TDM
    if (wave < HEADS) {
        const __bf16* src = kb + ((size_t)(b*HEADS + wave))*NPTS*16;
        tdm_copy_to_lds((unsigned)(size_t)(void*)&sK[wave][0], src, NPTS*16*2);
        __builtin_amdgcn_s_wait_tensorcnt((short)0);
    }
#else
    {
        const u32x4* src = reinterpret_cast<const u32x4*>(kb + (size_t)b*HEADS*NPTS*16);
        u32x4* dst = reinterpret_cast<u32x4*>(&sK[0][0]);
        for (int i = tid; i < (HEADS*NPTS*16*2)/16; i += 256) dst[i] = src[i];
    }
#endif
    for (int i = tid; i < 256; i += 256) { sQr[i] = qo[i]; sQi[i] = qo[256 + i]; }
    if (tid < 16) { float t = theta[tid]; sCos[tid] = cosf(t); sSin[tid] = sinf(t); }
    __syncthreads();

    const float scale = amp[0] * 0.35355339059327373f;   // amp / sqrt(8)
    const int   r     = lane & 15;
    const bool  hiL   = lane >= 16;
    const size_t bh   = (size_t)(b*HEADS + h);

    // A operands: lanes 0-15 hold K0..7 (re half), lanes 16-31 hold K8..15 (im half)
    const __bf16* qrow = qb + (bh*NPTS + rowBase + r)*16;
    const u32x4 SGN = {0x80008000u, 0x80008000u, 0x80008000u, 0x80008000u};
    const u32x4 Z4  = {0u, 0u, 0u, 0u};
    U16bf aRe, aIm;
    aRe.q[1] = Z4; aIm.q[1] = Z4;
    aRe.q[0] = *reinterpret_cast<const u32x4*>(qrow + (hiL ? 8 : 0));   // qr | qi
    {
        u32x4 t = *reinterpret_cast<const u32x4*>(qrow + (hiL ? 0 : 8)); // qi | qr
        if (hiL) t ^= SGN;                                               // qi | -qr
        aIm.q[0] = t;
    }

    float mrun[8], lrun[8];
    #pragma unroll
    for (int i = 0; i < 8; ++i) { mrun[i] = -1e30f; lrun[i] = 0.f; }
    v8f acc = {};

    const __bf16* kpan  = &sK[h][0];
    const __bf16* vbase = vb + bh*16*NPTS;
    __bf16* myP = &sP[wave][0];

    for (int j0 = 0; j0 < NPTS; j0 += 32) {
        // ---- score B operands from LDS K panel ----
        U16bf bk0, bk1;
        bk0.q[1] = Z4; bk1.q[1] = Z4;
        bk0.q[0] = *reinterpret_cast<const u32x4*>(kpan + (size_t)(j0 + r)*16      + (hiL ? 8 : 0));
        bk1.q[0] = *reinterpret_cast<const u32x4*>(kpan + (size_t)(j0 + 16 + r)*16 + (hiL ? 8 : 0));

        v8f z8 = {};
        v8f sre0 = __builtin_amdgcn_wmma_f32_16x16x32_bf16(false, aRe.v, false, bk0.v, (short)0, z8, false, false);
        v8f sim0 = __builtin_amdgcn_wmma_f32_16x16x32_bf16(false, aIm.v, false, bk0.v, (short)0, z8, false, false);
        v8f sre1 = __builtin_amdgcn_wmma_f32_16x16x32_bf16(false, aRe.v, false, bk1.v, (short)0, z8, false, false);
        v8f sim1 = __builtin_amdgcn_wmma_f32_16x16x32_bf16(false, aIm.v, false, bk1.v, (short)0, z8, false, false);

        // prefetch next V chunk into cache
        if (j0 + 32 < NPTS) __builtin_prefetch(vbase + (size_t)r*NPTS + j0 + 32, 0, 1);

        // ---- |score| and online softmax (row stats live per-reg, per 16-lane half) ----
        float p0[8], p1[8];
        #pragma unroll
        for (int i = 0; i < 8; ++i) {
            float m0 = sqrtf(sre0[i]*sre0[i] + sim0[i]*sim0[i]) * scale;
            float m1 = sqrtf(sre1[i]*sre1[i] + sim1[i]*sim1[i]) * scale;
            p0[i] = m0; p1[i] = m1;
            float cm = fmaxf(m0, m1);
            cm = fmaxf(cm, __shfl_xor(cm, 1, 32));
            cm = fmaxf(cm, __shfl_xor(cm, 2, 32));
            cm = fmaxf(cm, __shfl_xor(cm, 4, 32));
            cm = fmaxf(cm, __shfl_xor(cm, 8, 32));
            float nm = fmaxf(mrun[i], cm);
            float alpha = __expf(mrun[i] - nm);
            float e0 = __expf(p0[i] - nm);
            float e1 = __expf(p1[i] - nm);
            p0[i] = e0; p1[i] = e1;
            float rs = e0 + e1;
            rs += __shfl_xor(rs, 1, 32);
            rs += __shfl_xor(rs, 2, 32);
            rs += __shfl_xor(rs, 4, 32);
            rs += __shfl_xor(rs, 8, 32);
            lrun[i] = lrun[i]*alpha + rs;
            mrun[i] = nm;
            acc[i] *= alpha;
        }

        // ---- P: C layout -> LDS -> A layout ----
        #pragma unroll
        for (int i = 0; i < 8; ++i) {
            const int row = i + (hiL ? 8 : 0);
            myP[row*32 + r]      = (__bf16)p0[i];
            myP[row*32 + 16 + r] = (__bf16)p1[i];
        }
        U16bf aP;
        {
            const __bf16* prow = myP + (size_t)r*32 + (hiL ? 8 : 0);
            aP.q[0] = *reinterpret_cast<const u32x4*>(prow);
            aP.q[1] = *reinterpret_cast<const u32x4*>(prow + 16);
        }

        // ---- V B operand: lane = output comp, K = key index (full K=32) ----
        U16bf bV;
        {
            const __bf16* vrow = vbase + (size_t)r*NPTS + j0 + (hiL ? 8 : 0);
            bV.q[0] = *reinterpret_cast<const u32x4*>(vrow);
            bV.q[1] = *reinterpret_cast<const u32x4*>(vrow + 16);
        }
        acc = __builtin_amdgcn_wmma_f32_16x16x32_bf16(false, aP.v, false, bV.v, (short)0, acc, false, false);
    }

    // ---- finalize softmax, merge heads into LDS ----
    #pragma unroll
    for (int i = 0; i < 8; ++i) acc[i] *= (1.0f / lrun[i]);
    #pragma unroll
    for (int i = 0; i < 8; ++i) {
        const int row = qt*16 + i + (hiL ? 8 : 0);
        const int c   = r;                              // comp: 0..7 re, 8..15 im
        const int col = (c < 8) ? (h*8 + c) : (16 + h*8 + (c - 8));
        sComb[row][col] = acc[i];
    }
    __syncthreads();

    // ---- epilogue: phase gate, Wo unitary, re + 0.1*im ----
    for (int task = tid; task < 64*16; task += 256) {
        const int row = task >> 4;
        const int ep  = task & 15;
        float zr = 0.f, zi = 0.f;
        #pragma unroll
        for (int e = 0; e < 16; ++e) {
            const float yr = sComb[row][e], yi = sComb[row][16 + e];
            const float pr = yr*sCos[e] - yi*sSin[e];
            const float pi = yr*sSin[e] + yi*sCos[e];
            const float wr = sQr[ep*16 + e], wi = sQi[ep*16 + e];
            zr += pr*wr + pi*wi;        // y . conj(Qo[ep,:])
            zi += pi*wr - pr*wi;
        }
        const int n = tb*64 + row;
        Xout[((size_t)b*NPTS + n)*16 + ep] = zr + 0.1f*zi;
    }
}

// ---------------------------------------------------------------------------
// Kernel 4: out = x @ w_out.T + b_out
// ---------------------------------------------------------------------------
__global__ __launch_bounds__(256) void lin_out_kernel(const float* __restrict__ X,
                                                      const float* __restrict__ w,
                                                      const float* __restrict__ bv,
                                                      float* __restrict__ out, int odim) {
    const int row = blockIdx.x * 256 + threadIdx.x;
    float x[16];
    #pragma unroll
    for (int c = 0; c < 16; ++c) x[c] = X[(size_t)row*16 + c];
    for (int o = 0; o < odim; ++o) {
        float a = bv[o];
        #pragma unroll
        for (int c = 0; c < 16; ++c) a += x[c] * w[o*16 + c];
        out[(size_t)row*odim + o] = a;
    }
}

__device__ inline float blockReduceSum(float* red, float v, int tid) {
    red[tid] = v; __syncthreads();
    for (int s = 128; s > 0; s >>= 1) { if (tid < s) red[tid] += red[tid + s]; __syncthreads(); }
    float r = red[0]; __syncthreads();
    return r;
}

// ---------------------------------------------------------------------------
// Kernel 5: softmax(heat) over N; pos_pred = sum(xyz * w)
// ---------------------------------------------------------------------------
__global__ __launch_bounds__(256) void pos_kernel(const float* __restrict__ heat,
                                                  const float* __restrict__ xyz,
                                                  float* __restrict__ posp,
                                                  float* __restrict__ dout) {
    __shared__ float red[256];
    const int b = blockIdx.x, tid = threadIdx.x;
    const float* hb = heat + (size_t)b*NPTS;
    float mx = -1e30f;
    for (int n = tid; n < NPTS; n += 256) mx = fmaxf(mx, hb[n]);
    red[tid] = mx; __syncthreads();
    for (int s = 128; s > 0; s >>= 1) { if (tid < s) red[tid] = fmaxf(red[tid], red[tid + s]); __syncthreads(); }
    mx = red[0]; __syncthreads();
    float s0 = 0, s1 = 0, s2 = 0, s3 = 0;
    for (int n = tid; n < NPTS; n += 256) {
        const float e = __expf(hb[n] - mx);
        const float* p = xyz + ((size_t)b*NPTS + n)*3;
        s0 += e; s1 += e*p[0]; s2 += e*p[1]; s3 += e*p[2];
    }
    s0 = blockReduceSum(red, s0, tid);
    s1 = blockReduceSum(red, s1, tid);
    s2 = blockReduceSum(red, s2, tid);
    s3 = blockReduceSum(red, s3, tid);
    if (tid == 0) {
        const float inv = 1.0f / s0;
        const float px = s1*inv, py = s2*inv, pz = s3*inv;
        posp[b*3+0] = px; posp[b*3+1] = py; posp[b*3+2] = pz;
        dout[b*3+0] = px; dout[b*3+1] = py; dout[b*3+2] = pz;
    }
}

// ---------------------------------------------------------------------------
// Kernel 6: radius-masked mean of feats (fallback: global mean), row-normalize.
// ---------------------------------------------------------------------------
__global__ __launch_bounds__(256) void orient_kernel(const float* __restrict__ feats,
                                                     const float* __restrict__ xyz,
                                                     const float* __restrict__ posp,
                                                     float* __restrict__ dout) {
    __shared__ float red[256];
    __shared__ float sOut[19];
    const int b = blockIdx.x, tid = threadIdx.x;
    const float px = posp[b*3+0], py = posp[b*3+1], pz = posp[b*3+2];
    const float r2 = 0.02f * 0.02f;
    float cnt = 0.f, sel[9], tot[9];
    #pragma unroll
    for (int k = 0; k < 9; ++k) { sel[k] = 0.f; tot[k] = 0.f; }
    for (int n = tid; n < NPTS; n += 256) {
        const float* p = xyz + ((size_t)b*NPTS + n)*3;
        const float dx = p[0]-px, dy = p[1]-py, dz = p[2]-pz;
        const bool in = (dx*dx + dy*dy + dz*dz) < r2;
        const float* f = feats + ((size_t)b*NPTS + n)*9;
        #pragma unroll
        for (int k = 0; k < 9; ++k) {
            const float fv = f[k];
            tot[k] += fv;
            if (in) sel[k] += fv;
        }
        if (in) cnt += 1.f;
    }
    {
        float v = blockReduceSum(red, cnt, tid);
        if (tid == 0) sOut[0] = v;
    }
    #pragma unroll
    for (int k = 0; k < 9; ++k) {
        float v = blockReduceSum(red, sel[k], tid);
        if (tid == 0) sOut[1 + k] = v;
    }
    #pragma unroll
    for (int k = 0; k < 9; ++k) {
        float v = blockReduceSum(red, tot[k], tid);
        if (tid == 0) sOut[10 + k] = v;
    }
    __syncthreads();
    if (tid == 0) {
        const float c = sOut[0];
        float m[9];
        const float invSel = 1.0f / fmaxf(c, 1.0f);
        const float invTot = 1.0f / (float)NPTS;
        #pragma unroll
        for (int k = 0; k < 9; ++k)
            m[k] = (c > 0.f) ? sOut[1+k]*invSel : sOut[10+k]*invTot;
        #pragma unroll
        for (int rrow = 0; rrow < 3; ++rrow) {
            const float nr = sqrtf(m[rrow*3]*m[rrow*3] + m[rrow*3+1]*m[rrow*3+1] + m[rrow*3+2]*m[rrow*3+2]);
            const float inv = 1.0f / (nr + 1e-8f);
            #pragma unroll
            for (int cc = 0; cc < 3; ++cc)
                dout[24 + b*9 + rrow*3 + cc] = m[rrow*3 + cc] * inv;
        }
    }
}

// ---------------------------------------------------------------------------
// Host: input flattening follows jax tree order (dicts sorted by key):
//   0 xyz, 1 rgb,
//   pos_params: 2 b_in, 3 b_out,
//     layer l (base = 4 + 10*l): amp,theta,wk_i,wk_r,wo_i,wo_r,wq_i,wq_r,wv_i,wv_r
//     24 w_in, 25 w_out
//   ori_params: 26 b_in, 27 b_out, layers at 28/38, 48 w_in, 49 w_out
// ---------------------------------------------------------------------------
extern "C" void kernel_launch(void* const* d_in, const int* in_sizes, int n_in,
                              void* d_out, int out_size, void* d_ws, size_t ws_size,
                              hipStream_t stream) {
    (void)in_sizes; (void)n_in; (void)out_size; (void)ws_size;
    const float* xyz = (const float*)d_in[0];
    const float* rgb = (const float*)d_in[1];
    auto P = [&](int i) { return (const float*)d_in[i]; };
    auto layerBase = [&](int bb, int l) { return (bb ? 26 : 2) + 2 + l*10; };

    char* ws = (char*)d_ws;
    float*  QM    = (float*)(ws + WS_QMAT);
    float*  X0    = (float*)(ws + WS_X0);
    float*  X1    = (float*)(ws + WS_X1);
    __bf16* QB    = (__bf16*)(ws + WS_QB);
    __bf16* KB    = (__bf16*)(ws + WS_KB);
    __bf16* VB    = (__bf16*)(ws + WS_VB);
    float*  HEAT  = (float*)(ws + WS_HEAT);
    float*  FEATS = (float*)(ws + WS_FEATS);
    float*  POSP  = (float*)(ws + WS_POSP);
    float*  OUT   = (float*)d_out;

    // --- QR of all 16 unitaries ---
    QRArgs qa;
    for (int bb = 0; bb < 2; ++bb)
        for (int l = 0; l < 2; ++l) {
            const int base = layerBase(bb, l);
            const int m = (bb*2 + l)*4;
            qa.wr[m+0] = P(base+7); qa.wi[m+0] = P(base+6);   // q
            qa.wr[m+1] = P(base+3); qa.wi[m+1] = P(base+2);   // k
            qa.wr[m+2] = P(base+9); qa.wi[m+2] = P(base+8);   // v
            qa.wr[m+3] = P(base+5); qa.wi[m+3] = P(base+4);   // o
        }
    qr16_kernel<<<16, 32, 0, stream>>>(qa, QM);

    const int ROWS = BATCH * NPTS;
    for (int bb = 0; bb < 2; ++bb) {
        const float* w_in  = P(bb ? 48 : 24);
        const float* b_in  = P(bb ? 26 : 2);
        const float* w_out = P(bb ? 49 : 25);
        const float* b_out = P(bb ? 27 : 3);

        lin_in_kernel<<<ROWS/256, 256, 0, stream>>>(rgb, w_in, b_in, X0);
        float* cur = X0;
        float* nxt = X1;
        for (int l = 0; l < 2; ++l) {
            const int m = (bb*2 + l)*4;
            const int base = layerBase(bb, l);
            proj_qkv_kernel<<<ROWS/256, 256, 0, stream>>>(
                cur, QM + (size_t)m*512, QM + (size_t)(m+1)*512, QM + (size_t)(m+2)*512,
                QB, KB, VB);
            attn_kernel<<<BATCH*(NPTS/64), 256, 0, stream>>>(
                QB, KB, VB, QM + (size_t)(m+3)*512, P(base+1), P(base+0), nxt);
            float* t = cur; cur = nxt; nxt = t;
        }
        if (bb == 0)
            lin_out_kernel<<<ROWS/256, 256, 0, stream>>>(cur, w_out, b_out, HEAT, 1);
        else
            lin_out_kernel<<<ROWS/256, 256, 0, stream>>>(cur, w_out, b_out, FEATS, 9);
    }

    pos_kernel<<<BATCH, 256, 0, stream>>>(HEAT, xyz, POSP, OUT);
    orient_kernel<<<BATCH, 256, 0, stream>>>(FEATS, xyz, POSP, OUT);
}
